// EncoderLayer_9448928051629
// MI455X (gfx1250) — compile-verified, matrix operand order checked by reading, not run on hardware
//
#include <hip/hip_runtime.h>
#include <hip/hip_bf16.h>

// ---------------------------------------------------------------------------
// Transformer encoder layer for MI455X (gfx1250), bf16 WMMA + f32 accumulate.
// B=2, S=2048, D=1024, H=16, Dh=64, DFF=4096, M = B*S = 4096.
// ---------------------------------------------------------------------------

typedef __bf16 bf16;
typedef __attribute__((ext_vector_type(16))) bf16  v16bf;
typedef __attribute__((ext_vector_type(8)))  bf16  bf16x8;
typedef __attribute__((ext_vector_type(8)))  float v8f;

#define S_LEN   2048
#define D_MODEL 1024
#define N_HEADS 16
#define HEAD_D  64
#define D_FF    4096
#define M_ROWS  4096

__device__ __forceinline__ v8f vzero8() {
  v8f z = {0.f,0.f,0.f,0.f,0.f,0.f,0.f,0.f};
  return z;
}

// Wait for all outstanding async (memory->LDS) transfers from this wave.
__device__ __forceinline__ void wait_async0() {
#if __has_builtin(__builtin_amdgcn_s_wait_asynccnt)
  __builtin_amdgcn_s_wait_asynccnt(0);
#else
  asm volatile("s_wait_asynccnt 0x0" ::: "memory");
#endif
}

// Issue GLOBAL_LOAD_ASYNC_TO_LDS_B128 (GVS mode): mem = saddr + voff,
// LDS dest = low 32 bits of flat shared address (LDS aperture, ISA 10.2).
__device__ __forceinline__ void async_copy16(void* lds_dst,
                                             unsigned long long saddr,
                                             unsigned voff_bytes) {
  unsigned la = (unsigned)(uintptr_t)lds_dst;
  asm volatile("global_load_async_to_lds_b128 %0, %1, %2"
               :: "v"(la), "v"(voff_bytes), "s"(saddr) : "memory");
}

// Load one 16x32 bf16 operand fragment. Per ISA 16-bit A layout:
// lane L: row = L%16, elems 0-7 <-> K = 8*(L/16)+0..7, elems 8-15 <-> 16+8*(L/16)+0..7
// => two contiguous 16-byte chunks per lane. Same pattern serves B-tiles
// loaded from a [N][K] (pre-transposed) matrix.
__device__ __forceinline__ v16bf load_frag(const bf16* rowp, int hf) {
  union { v16bf v; bf16x8 h[2]; } u;
  u.h[0] = *(const bf16x8*)(rowp + hf * 8);
  u.h[1] = *(const bf16x8*)(rowp + 16 + hf * 8);
  return u.v;
}

__device__ __forceinline__ v8f wmma_bf16(v16bf a, v16bf b, v8f c) {
  return __builtin_amdgcn_wmma_f32_16x16x32_bf16(false, a, false, b,
                                                 (short)0, c, false, false);
}

// ---------------------------------------------------------------------------
// Elementwise f32 -> bf16 convert (4 elems/thread, float4 loads)
// ---------------------------------------------------------------------------
__global__ __launch_bounds__(256) void cvt_bf16_kernel(
    const float* __restrict__ in, bf16* __restrict__ out, int n4) {
  int i = blockIdx.x * blockDim.x + threadIdx.x;
  if (i < n4) {
    float4 v = ((const float4*)in)[i];
    out[4 * i + 0] = (bf16)v.x;
    out[4 * i + 1] = (bf16)v.y;
    out[4 * i + 2] = (bf16)v.z;
    out[4 * i + 3] = (bf16)v.w;
  }
}

// ---------------------------------------------------------------------------
// Transpose + convert: w [K][N] f32  ->  wT [N][K] bf16  (32x32 LDS tiles)
// ---------------------------------------------------------------------------
__global__ __launch_bounds__(256) void transpose_cvt_kernel(
    const float* __restrict__ w, bf16* __restrict__ wT, int K, int N) {
  __shared__ float tile[32][33];
  int nx = blockIdx.x * 32, ky = blockIdx.y * 32;
  int tx = threadIdx.x, ty = threadIdx.y;   // blockDim = (32, 8)
#pragma unroll
  for (int j = 0; j < 4; j++)
    tile[ty + j * 8][tx] = w[(size_t)(ky + ty + j * 8) * N + nx + tx];
  __syncthreads();
#pragma unroll
  for (int j = 0; j < 4; j++)
    wT[(size_t)(nx + ty + j * 8) * K + ky + tx] = (bf16)tile[tx][ty + j * 8];
}

// ---------------------------------------------------------------------------
// GEMM: out[M][N] = A[M][K](bf16) * Bt[N][K](bf16)^T + bias, epilogue variants.
// Macro tile 128x128, 8 waves (4 M x 2 N), wave tile 32x64 (2x4 accumulators),
// K-step 32, LDS double-buffered via GLOBAL_LOAD_ASYNC_TO_LDS_B128.
// ---------------------------------------------------------------------------
enum { EPI_BF16 = 0, EPI_VT = 1, EPI_RELU = 2, EPI_RESID = 3 };

template <int EPI>
__global__ __launch_bounds__(256) void gemm_bf16_kernel(
    const bf16* __restrict__ A, const bf16* __restrict__ Bt,
    const float* __restrict__ bias, const float* __restrict__ resid,
    void* __restrict__ outv, int M, int N, int K) {
  __shared__ __align__(16) bf16 lsA[2][128][40];
  __shared__ __align__(16) bf16 lsB[2][128][40];

  const int t    = threadIdx.x;
  const int wid  = t >> 5;
  const int lane = t & 31;
  const int l16  = lane & 15;
  const int hf   = lane >> 4;
  const int wm   = wid & 3;   // M quadrant (32 rows)
  const int wn   = wid >> 2;  // N half (64 cols)
  const int m0   = blockIdx.y * 128;
  const int n0   = blockIdx.x * 128;

  const unsigned long long Abase = (unsigned long long)(uintptr_t)A;
  const unsigned long long Bbase = (unsigned long long)(uintptr_t)Bt;

  v8f acc[2][4];
#pragma unroll
  for (int i = 0; i < 2; i++)
#pragma unroll
    for (int j = 0; j < 4; j++) acc[i][j] = vzero8();

  // async stage one 128x32 A tile + one 128x32 B tile into LDS buffer `buf`
  auto loadTile = [&](int buf, int k0) {
#pragma unroll
    for (int i = 0; i < 2; i++) {
      int c   = t + i * 256;       // 0..511 chunks of 8 bf16 (16B)
      int row = c >> 2;
      int cg  = (c & 3) * 8;
      unsigned offA = ((unsigned)(m0 + row) * (unsigned)K + (unsigned)(k0 + cg)) * 2u;
      unsigned offB = ((unsigned)(n0 + row) * (unsigned)K + (unsigned)(k0 + cg)) * 2u;
      async_copy16(&lsA[buf][row][cg], Abase, offA);
      async_copy16(&lsB[buf][row][cg], Bbase, offB);
    }
  };

  const int nk = K >> 5;
  loadTile(0, 0);
  wait_async0();
  __syncthreads();

  for (int ks = 0; ks < nk; ks++) {
    int buf = ks & 1;
    if (ks + 1 < nk) loadTile(buf ^ 1, (ks + 1) * 32);

    v16bf af[2], bfr[4];
#pragma unroll
    for (int mi = 0; mi < 2; mi++)
      af[mi] = load_frag(&lsA[buf][wm * 32 + mi * 16 + l16][0], hf);
#pragma unroll
    for (int ni = 0; ni < 4; ni++)
      bfr[ni] = load_frag(&lsB[buf][wn * 64 + ni * 16 + l16][0], hf);
#pragma unroll
    for (int mi = 0; mi < 2; mi++)
#pragma unroll
      for (int ni = 0; ni < 4; ni++)
        acc[mi][ni] = wmma_bf16(af[mi], bfr[ni], acc[mi][ni]);

    wait_async0();      // next-buffer async copies complete (after compute)
    __syncthreads();
  }

  // Epilogue. C layout: lane holds col n = l16 (+tile), VGPR i -> row i + hf*8.
#pragma unroll
  for (int mi = 0; mi < 2; mi++) {
#pragma unroll
    for (int ni = 0; ni < 4; ni++) {
      int n    = n0 + wn * 64 + ni * 16 + l16;
      float bb = bias[n];
      v8f a    = acc[mi][ni];
#pragma unroll
      for (int i = 0; i < 8; i++) {
        int m   = m0 + wm * 32 + mi * 16 + hf * 8 + i;
        float v = a[i] + bb;
        if (EPI == EPI_RELU) v = v > 0.f ? v : 0.f;
        if (EPI == EPI_RESID) {
          v += resid[(size_t)m * N + n];
          ((float*)outv)[(size_t)m * N + n] = v;
        } else if (EPI == EPI_VT) {
          // store V transposed per batch: vT[b][n][s], b = m/S, s = m%S
          int b = m >> 11, s = m & (S_LEN - 1);
          ((bf16*)outv)[((size_t)b * D_MODEL + n) * S_LEN + s] = (bf16)v;
        } else {
          ((bf16*)outv)[(size_t)m * N + n] = (bf16)v;
        }
      }
    }
  }
}

// ---------------------------------------------------------------------------
// Flash attention: one wave per (b, h, 16-query tile). KV chunks of 64.
// Row-sum of P done with a WMMA against an all-ones B matrix (each column of
// D holds the row sum). qb/kb: bf16 [B*S][D]; vT: bf16 [B][D][S].
// ---------------------------------------------------------------------------
__global__ __launch_bounds__(256) void attn_kernel(
    const bf16* __restrict__ qb, const bf16* __restrict__ kb,
    const bf16* __restrict__ vT, const int* __restrict__ mask,
    bf16* __restrict__ ob) {
  __shared__ __align__(16) bf16 pst[8][16][72];  // per-wave P staging (16x64)

  const int wid  = threadIdx.x >> 5;
  const int lane = threadIdx.x & 31;
  const int l16  = lane & 15;
  const int hf   = lane >> 4;

  int gw  = blockIdx.x * 8 + wid;  // 0..4095
  int b   = gw >> 11;
  int rem = gw & 2047;
  int h   = rem >> 7;
  int mt  = rem & 127;

  // Q fragments (held across the whole kv loop)
  const bf16* Qrow =
      qb + ((size_t)(b * S_LEN + mt * 16 + l16)) * D_MODEL + h * HEAD_D;
  v16bf q0 = load_frag(Qrow, hf);       // Dh 0..31
  v16bf q1 = load_frag(Qrow + 32, hf);  // Dh 32..63

  // all-ones B operand for WMMA row-sum
  union { v16bf v; bf16 e[16]; } uo;
#pragma unroll
  for (int e = 0; e < 16; e++) uo.e[e] = (bf16)1.0f;
  const v16bf ones = uo.v;

  float rm[8], rs[8];
#pragma unroll
  for (int i = 0; i < 8; i++) { rm[i] = -1e30f; rs[i] = 0.f; }
  v8f o[4];
#pragma unroll
  for (int ni = 0; ni < 4; ni++) o[ni] = vzero8();

  const float sc = 0.125f;  // 1/sqrt(64)

  for (int kc = 0; kc < S_LEN / 64; kc++) {
    // ---- scores S = Q * K^T for 64 kv positions (four 16-col tiles) ----
    const bf16* Krow =
        kb + ((size_t)(b * S_LEN + kc * 64 + l16)) * D_MODEL + h * HEAD_D;
    v8f s[4];
#pragma unroll
    for (int tt = 0; tt < 4; tt++) {
      v16bf k0 = load_frag(Krow + (size_t)tt * 16 * D_MODEL, hf);
      v16bf k1 = load_frag(Krow + (size_t)tt * 16 * D_MODEL + 32, hf);
      s[tt] = wmma_bf16(q0, k0, vzero8());
      s[tt] = wmma_bf16(q1, k1, s[tt]);
    }
    float madd[4];
#pragma unroll
    for (int tt = 0; tt < 4; tt++)
      madd[tt] =
          (mask[b * S_LEN + kc * 64 + tt * 16 + l16] == 0) ? -1e9f : 0.f;

    // ---- online softmax: max via lane shuffles, exp, stage P to LDS ----
#pragma unroll
    for (int i = 0; i < 8; i++) {
      float a0 = s[0][i] * sc + madd[0];
      float a1 = s[1][i] * sc + madd[1];
      float a2 = s[2][i] * sc + madd[2];
      float a3 = s[3][i] * sc + madd[3];
      float cm = fmaxf(fmaxf(a0, a1), fmaxf(a2, a3));
#pragma unroll
      for (int x = 1; x <= 8; x <<= 1) cm = fmaxf(cm, __shfl_xor(cm, x, 32));
      float mn   = fmaxf(rm[i], cm);
      float corr = __expf(rm[i] - mn);
      rm[i] = mn;
      rs[i] *= corr;
#pragma unroll
      for (int ni = 0; ni < 4; ni++) o[ni][i] *= corr;
      int r = i + hf * 8;
      pst[wid][r][l16]      = (bf16)__expf(a0 - mn);
      pst[wid][r][16 + l16] = (bf16)__expf(a1 - mn);
      pst[wid][r][32 + l16] = (bf16)__expf(a2 - mn);
      pst[wid][r][48 + l16] = (bf16)__expf(a3 - mn);
    }

    // ---- reload P as A-operand fragments ----
    v16bf pf0 = load_frag(&pst[wid][l16][0], hf);   // kv 0..31 of chunk
    v16bf pf1 = load_frag(&pst[wid][l16][32], hf);  // kv 32..63 of chunk

    // row-sum of P via WMMA with ones (replaces lane-shuffle sum reduction)
    v8f srow = wmma_bf16(pf0, ones, vzero8());
    srow     = wmma_bf16(pf1, ones, srow);
#pragma unroll
    for (int i = 0; i < 8; i++) rs[i] += srow[i];

    // ---- O += P * V  (B operand from vT [b][d][s]) ----
    const bf16* Vrow =
        vT + ((size_t)b * D_MODEL + h * HEAD_D + l16) * S_LEN + kc * 64;
#pragma unroll
    for (int ni = 0; ni < 4; ni++) {
      v16bf vf0 = load_frag(Vrow + (size_t)ni * 16 * S_LEN, hf);
      v16bf vf1 = load_frag(Vrow + (size_t)ni * 16 * S_LEN + 32, hf);
      o[ni] = wmma_bf16(pf0, vf0, o[ni]);
      o[ni] = wmma_bf16(pf1, vf1, o[ni]);
    }
  }

  // ---- normalize and store ----
#pragma unroll
  for (int ni = 0; ni < 4; ni++) {
#pragma unroll
    for (int i = 0; i < 8; i++) {
      float v  = o[ni][i] / rs[i];
      size_t m = (size_t)(b * S_LEN + mt * 16 + hf * 8 + i);
      ob[m * D_MODEL + h * HEAD_D + ni * 16 + l16] = (bf16)v;
    }
  }
}

// ---------------------------------------------------------------------------
// LayerNorm: one 256-thread block per row of 1024. Emits f32 (+ optional bf16).
// ---------------------------------------------------------------------------
__global__ __launch_bounds__(256) void ln_kernel(
    const float* __restrict__ in, const float* __restrict__ g,
    const float* __restrict__ be, float* __restrict__ outf,
    bf16* __restrict__ outb) {
  __shared__ float shs[8], shs2[8], fin[2];
  const int row  = blockIdx.x;
  const int t    = threadIdx.x;
  const int wid  = t >> 5;
  const int lane = t & 31;
  const float* p = in + (size_t)row * D_MODEL;

  float v[4], s = 0.f, s2 = 0.f;
#pragma unroll
  for (int j = 0; j < 4; j++) {
    v[j] = p[t + j * 256];
    s += v[j];
    s2 += v[j] * v[j];
  }
#pragma unroll
  for (int x = 16; x >= 1; x >>= 1) {
    s += __shfl_xor(s, x, 32);
    s2 += __shfl_xor(s2, x, 32);
  }
  if (lane == 0) { shs[wid] = s; shs2[wid] = s2; }
  __syncthreads();
  if (t == 0) {
    float a = 0.f, a2 = 0.f;
#pragma unroll
    for (int k = 0; k < 8; k++) { a += shs[k]; a2 += shs2[k]; }
    fin[0] = a; fin[1] = a2;
  }
  __syncthreads();
  const float inv_n = 1.0f / (float)D_MODEL;
  float mu  = fin[0] * inv_n;
  float var = fin[1] * inv_n - mu * mu;
  float inv = rsqrtf(var + 1e-5f);
#pragma unroll
  for (int j = 0; j < 4; j++) {
    int c   = t + j * 256;
    float y = (v[j] - mu) * inv * g[c] + be[c];
    outf[(size_t)row * D_MODEL + c] = y;
    if (outb) outb[(size_t)row * D_MODEL + c] = (bf16)y;
  }
}

// ---------------------------------------------------------------------------
// Host launcher
// ---------------------------------------------------------------------------
extern "C" void kernel_launch(void* const* d_in, const int* in_sizes, int n_in,
                              void* d_out, int out_size, void* d_ws,
                              size_t ws_size, hipStream_t stream) {
  const float* x    = (const float*)d_in[0];
  const int* mask   = (const int*)d_in[1];
  const float* wq   = (const float*)d_in[2];
  const float* bq   = (const float*)d_in[3];
  const float* wk   = (const float*)d_in[4];
  const float* bk   = (const float*)d_in[5];
  const float* wv   = (const float*)d_in[6];
  const float* bv   = (const float*)d_in[7];
  const float* wo   = (const float*)d_in[8];
  const float* bo   = (const float*)d_in[9];
  const float* w1   = (const float*)d_in[10];
  const float* b1   = (const float*)d_in[11];
  const float* w2   = (const float*)d_in[12];
  const float* b2   = (const float*)d_in[13];
  const float* ln1g = (const float*)d_in[14];
  const float* ln1b = (const float*)d_in[15];
  const float* ln2g = (const float*)d_in[16];
  const float* ln2b = (const float*)d_in[17];

  char* ws        = (char*)d_ws;
  const size_t MB = (size_t)1 << 20;
  bf16* xb   = (bf16*)(ws + 0 * MB);    // 8 MB  [M][D]
  bf16* qb   = (bf16*)(ws + 8 * MB);    // 8 MB
  bf16* kbuf = (bf16*)(ws + 16 * MB);   // 8 MB
  bf16* vT   = (bf16*)(ws + 24 * MB);   // 8 MB  [B][D][S]
  bf16* obuf = (bf16*)(ws + 32 * MB);   // 8 MB
  bf16* wqT  = (bf16*)(ws + 40 * MB);   // 2 MB each
  bf16* wkT  = (bf16*)(ws + 42 * MB);
  bf16* wvT  = (bf16*)(ws + 44 * MB);
  bf16* woT  = (bf16*)(ws + 46 * MB);
  bf16* w1T  = (bf16*)(ws + 48 * MB);   // 8 MB  [DFF][D]
  bf16* w2T  = (bf16*)(ws + 56 * MB);   // 8 MB  [D][DFF]
  float* y1  = (float*)(ws + 64 * MB);  // 16 MB [M][D]  x + attn_out
  float* x1f = (float*)(ws + 80 * MB);  // 16 MB post-LN1 f32
  bf16* x1b  = (bf16*)(ws + 96 * MB);   // 8 MB  post-LN1 bf16
  bf16* hb   = (bf16*)(ws + 8 * MB);    // 32 MB [M][DFF], reuses q/k/v/o
  float* y2  = (float*)(ws + 104 * MB); // 16 MB pre-LN2

  dim3 blk256(256);
  dim3 tb(32, 8);

  // 1. convert activations
  cvt_bf16_kernel<<<(M_ROWS * D_MODEL / 4 + 255) / 256, blk256, 0, stream>>>(
      x, xb, M_ROWS * D_MODEL / 4);

  // 2. transpose+convert weights to [N][K] bf16
  transpose_cvt_kernel<<<dim3(D_MODEL / 32, D_MODEL / 32), tb, 0, stream>>>(
      wq, wqT, D_MODEL, D_MODEL);
  transpose_cvt_kernel<<<dim3(D_MODEL / 32, D_MODEL / 32), tb, 0, stream>>>(
      wk, wkT, D_MODEL, D_MODEL);
  transpose_cvt_kernel<<<dim3(D_MODEL / 32, D_MODEL / 32), tb, 0, stream>>>(
      wv, wvT, D_MODEL, D_MODEL);
  transpose_cvt_kernel<<<dim3(D_MODEL / 32, D_MODEL / 32), tb, 0, stream>>>(
      wo, woT, D_MODEL, D_MODEL);
  transpose_cvt_kernel<<<dim3(D_FF / 32, D_MODEL / 32), tb, 0, stream>>>(
      w1, w1T, D_MODEL, D_FF);
  transpose_cvt_kernel<<<dim3(D_MODEL / 32, D_FF / 32), tb, 0, stream>>>(
      w2, w2T, D_FF, D_MODEL);

  // 3. Q, K, V projections
  gemm_bf16_kernel<EPI_BF16><<<dim3(8, 32), blk256, 0, stream>>>(
      xb, wqT, bq, nullptr, qb, M_ROWS, D_MODEL, D_MODEL);
  gemm_bf16_kernel<EPI_BF16><<<dim3(8, 32), blk256, 0, stream>>>(
      xb, wkT, bk, nullptr, kbuf, M_ROWS, D_MODEL, D_MODEL);
  gemm_bf16_kernel<EPI_VT><<<dim3(8, 32), blk256, 0, stream>>>(
      xb, wvT, bv, nullptr, vT, M_ROWS, D_MODEL, D_MODEL);

  // 4. flash attention
  attn_kernel<<<dim3(M_ROWS / 8), blk256, 0, stream>>>(qb, kbuf, vT, mask,
                                                       obuf);

  // 5. output projection + residual (f32)
  gemm_bf16_kernel<EPI_RESID><<<dim3(8, 32), blk256, 0, stream>>>(
      obuf, woT, bo, x, y1, M_ROWS, D_MODEL, D_MODEL);

  // 6. LN1 -> f32 + bf16
  ln_kernel<<<dim3(M_ROWS), blk256, 0, stream>>>(y1, ln1g, ln1b, x1f, x1b);

  // 7. FFN1 (relu)
  gemm_bf16_kernel<EPI_RELU><<<dim3(32, 32), blk256, 0, stream>>>(
      x1b, w1T, b1, nullptr, hb, M_ROWS, D_FF, D_MODEL);

  // 8. FFN2 + residual (f32)
  gemm_bf16_kernel<EPI_RESID><<<dim3(8, 32), blk256, 0, stream>>>(
      hb, w2T, b2, x1f, y2, M_ROWS, D_MODEL, D_FF);

  // 9. LN2 -> d_out (f32)
  ln_kernel<<<dim3(M_ROWS), blk256, 0, stream>>>(y2, ln2g, ln2b, (float*)d_out,
                                                 nullptr);
}
